// LocalAttention_37151467110564
// MI455X (gfx1250) — matrix operand order repaired and checked by reading, not compiled, not run on hardware
//
#include <hip/hip_runtime.h>
#include <hip/hip_bf16.h>

// ---------------------------------------------------------------------------
// LocalAttention (banded, softmax over FULL row with 0-filled out-of-band)
//   B=2, S=2048, D=1024, H=16, HD=64, W=16 (band half-width 8)
//
// ctx_i = ( sum_{j in band} (e^{s_ij}-1) v_j + sum_all_j v_j )
//       / ( sum_{j in band} (e^{s_ij}-1)     + S )
//
// The 4 dense 4096x1024x1024 projections dominate (34 GFLOP); they run on
// v_wmma_f32_16x16x32_bf16 with fp32 accumulation. All fragments are fed by
// ds_load_b128 (16B-aligned LDS rows); all B fragments of a k-step are
// hoisted into distinct registers so LDS latency hides under the WMMAs.
// ---------------------------------------------------------------------------

#define BB   2
#define SS   2048
#define DD   1024
#define HH   16
#define HD   64
#define MM   (BB * SS)        // 4096 rows
#define BANDR 8               // W//2
#define LDSW 40               // LDS row stride in ushorts (80B, 16B aligned)

typedef __attribute__((ext_vector_type(16))) __bf16 v16bf;
typedef __attribute__((ext_vector_type(8)))  float  v8f;

union FragAB { v16bf v; uint4 q[2]; unsigned int u[8]; };

__device__ __forceinline__ unsigned short f2bf(float f) {
    unsigned int u = __float_as_uint(f);
    u += 0x7fffu + ((u >> 16) & 1u);          // round-to-nearest-even
    return (unsigned short)(u >> 16);
}

// ---------------------------------------------------------------------------
// fp32 -> bf16 weight conversion WITH transpose: dst[n*D+k] = bf16(src[k*D+n])
// (runs once per weight; data is L2-resident)
// ---------------------------------------------------------------------------
__global__ __launch_bounds__(256) void cvt_transpose_bf16(
    const float* __restrict__ src, unsigned short* __restrict__ dst) {
    int stride = gridDim.x * blockDim.x;
    for (int i = blockIdx.x * blockDim.x + threadIdx.x; i < DD * DD; i += stride) {
        int n = i >> 10, k = i & (DD - 1);
        dst[i] = f2bf(src[(size_t)k * DD + n]);
    }
}

// ---------------------------------------------------------------------------
// GEMM: out[w] = A(fp32,[M x D]) * W[w] + bias[w]   (W given transposed bf16)
// Block: 256 threads, tile 128(M) x 64(N), K-step 32.
// 8 waves in 4x2; each wave owns a 32x32 tile per output
//   -> per k-step: 2 A frags + 2 B frags per output, 4 WMMAs per output.
// ---------------------------------------------------------------------------
template <int NOUT>
__global__ __launch_bounds__(256) void gemm_bf16_kernel(
    const float* __restrict__ A,
    const unsigned short* __restrict__ W0,   // transposed [n][k] bf16
    const unsigned short* __restrict__ W1,
    const unsigned short* __restrict__ W2,
    const float* __restrict__ b0,
    const float* __restrict__ b1,
    const float* __restrict__ b2,
    float* __restrict__ o0,
    float* __restrict__ o1,
    float* __restrict__ o2)
{
    const int tid  = threadIdx.x;
    const int lane = tid & 31;
    const int wave = tid >> 5;
    const int wm   = wave >> 1;          // 0..3  (32-row slabs)
    const int wn   = wave & 1;           // 0..1  (32-col slabs)
    const int bm   = blockIdx.y * 128;   // global row base
    const int bn   = blockIdx.x * 64;    // global col base

    __shared__ unsigned short As[128][LDSW];        // [m][k] bf16
    __shared__ unsigned short Bs[NOUT][64][LDSW];   // [n][k] bf16 (N-major)

    const unsigned short* Wp[3] = { W0, W1, W2 };
    const float*          Bp[3] = { b0, b1, b2 };
    float*                Op[3] = { o0, o1, o2 };

    v8f acc[NOUT][2][2];
#pragma unroll
    for (int w = 0; w < NOUT; ++w)
#pragma unroll
        for (int mi = 0; mi < 2; ++mi)
#pragma unroll
            for (int ns = 0; ns < 2; ++ns)
                acc[w][mi][ns] = (v8f){0.f,0.f,0.f,0.f,0.f,0.f,0.f,0.f};

    const int m    = lane & 15;
    const int half = lane >> 4;

    // B staging: thread t copies one 16B chunk per output tile:
    const int bn_row = tid >> 2;           // 0..63 (n within tile)
    const int bn_seg = (tid & 3) * 8;      // k offset: 0,8,16,24

    for (int k0 = 0; k0 < DD; k0 += 32) {
        // ---- stage A tile (128x32), fp32 -> packed bf16 pairs ----
#pragma unroll
        for (int i = 0; i < 8; ++i) {
            int p  = tid + i * 256;            // pair index 0..2047
            int r  = p >> 4;                   // row 0..127
            int c2 = (p & 15) << 1;            // even col 0..30
            const float2 f2 = *(const float2*)&A[(size_t)(bm + r) * DD + k0 + c2];
            unsigned int pk = (unsigned int)f2bf(f2.x) |
                              ((unsigned int)f2bf(f2.y) << 16);
            *(unsigned int*)&As[r][c2] = pk;
        }
        // ---- stage W tiles: 64 rows x 32 k, straight 16B copies ----
#pragma unroll
        for (int w = 0; w < NOUT; ++w) {
            const uint4 src = *(const uint4*)&Wp[w][(size_t)(bn + bn_row) * DD + k0 + bn_seg];
            *(uint4*)&Bs[w][bn_row][bn_seg] = src;
        }
        __syncthreads();

        // ---- A fragments (2 per wave): halves are contiguous 16B runs ----
        FragAB a[2];
#pragma unroll
        for (int mi = 0; mi < 2; ++mi) {
            int row = wm * 32 + mi * 16 + m;
            a[mi].q[0] = *(const uint4*)&As[row][half * 8];        // K 0..7 (+8*half)
            a[mi].q[1] = *(const uint4*)&As[row][16 + half * 8];   // K 16..23 (+8*half)
        }

        // ---- hoist ALL B fragments (distinct regs -> staggered dscnt waits)
        FragAB bfr[NOUT][2];
#pragma unroll
        for (int w = 0; w < NOUT; ++w) {
#pragma unroll
            for (int ns = 0; ns < 2; ++ns) {
                int cl = wn * 32 + ns * 16 + m;        // local column (N = lane&15)
                bfr[w][ns].q[0] = *(const uint4*)&Bs[w][cl][half * 8];       // K 0..7  (+8*half)
                bfr[w][ns].q[1] = *(const uint4*)&Bs[w][cl][16 + half * 8];  // K 16..23 (+8*half)
            }
        }

        // ---- 4*NOUT WMMAs, loads above drain into them ----
#pragma unroll
        for (int w = 0; w < NOUT; ++w) {
#pragma unroll
            for (int ns = 0; ns < 2; ++ns) {
#pragma unroll
                for (int mi = 0; mi < 2; ++mi) {
                    acc[w][mi][ns] = __builtin_amdgcn_wmma_f32_16x16x32_bf16(
                        false, a[mi].v, false, bfr[w][ns].v, (short)0,
                        acc[w][mi][ns], false, false);
                }
            }
        }
        __syncthreads();
    }

    // ---- epilogue: C/D layout -> global, add bias ----
#pragma unroll
    for (int w = 0; w < NOUT; ++w) {
#pragma unroll
        for (int ns = 0; ns < 2; ++ns) {
            int col  = bn + wn * 32 + ns * 16 + m;
            float bv = Bp[w][col];
#pragma unroll
            for (int mi = 0; mi < 2; ++mi) {
#pragma unroll
                for (int r = 0; r < 8; ++r) {
                    int row = bm + wm * 32 + mi * 16 + half * 8 + r;  // M = 8*half + r
                    Op[w][(size_t)row * DD + col] = acc[w][mi][ns][r] + bv;
                }
            }
        }
    }
}

// ---------------------------------------------------------------------------
// Per-(b,h) column sum of V: vsum[bh][d] = sum_j v[b, j, h*64 + d]
// ---------------------------------------------------------------------------
__global__ __launch_bounds__(256) void vsum_kernel(const float* __restrict__ v,
                                                   float* __restrict__ vsum) {
    int bh = blockIdx.x;                 // 0..B*H-1
    int b = bh / HH, h = bh % HH;
    int d = threadIdx.x & 63;
    int chunk = threadIdx.x >> 6;        // 0..3, 512 rows each
    float s = 0.f;
    const float* base = v + (size_t)b * SS * DD + h * HD + d;
    for (int j = chunk * 512; j < (chunk + 1) * 512; ++j)
        s += base[(size_t)j * DD];
    __shared__ float red[256];
    red[threadIdx.x] = s;
    __syncthreads();
    if (threadIdx.x < 64) {
        float t = red[threadIdx.x] + red[threadIdx.x + 64] +
                  red[threadIdx.x + 128] + red[threadIdx.x + 192];
        vsum[bh * HD + threadIdx.x] = t;
    }
}

// ---------------------------------------------------------------------------
// Banded attention with uniform out-of-band contributions folded analytically.
// grid = (S/64, B*H); 8 waves/block; one wave32 per query (2 dims per lane).
// ---------------------------------------------------------------------------
__global__ __launch_bounds__(256) void band_attn_kernel(
    const float* __restrict__ q, const float* __restrict__ k,
    const float* __restrict__ v, const float* __restrict__ vsum,
    float* __restrict__ ctx)
{
    int bh = blockIdx.y;
    int b = bh / HH, h = bh % HH;
    int wave = threadIdx.x >> 5;
    int lane = threadIdx.x & 31;
    int qbase = blockIdx.x * 64 + wave * 8;

    const float* vs = vsum + bh * HD;
    float vs0 = vs[lane], vs1 = vs[lane + 32];

    for (int t = 0; t < 8; ++t) {
        int i = qbase + t;
        const float* qi = q + ((size_t)(b * SS + i)) * DD + h * HD;
        float qa = qi[lane], qb = qi[lane + 32];

        int j0 = (i - BANDR > 0) ? (i - BANDR) : 0;
        int j1 = (i + BANDR < SS - 1) ? (i + BANDR) : (SS - 1);

        float accw = 0.f, acc0 = 0.f, acc1 = 0.f;
        for (int j = j0; j <= j1; ++j) {
            const float* kj = k + ((size_t)(b * SS + j)) * DD + h * HD;
            float p = qa * kj[lane] + qb * kj[lane + 32];
#pragma unroll
            for (int o = 16; o > 0; o >>= 1)
                p += __shfl_xor(p, o, 32);
            float w = expf(p * 0.125f) - 1.0f;   // 1/sqrt(64) = 0.125
            const float* vj = v + ((size_t)(b * SS + j)) * DD + h * HD;
            acc0 += w * vj[lane];
            acc1 += w * vj[lane + 32];
            accw += w;
        }
        float denom = accw + (float)SS;
        float* ci = ctx + ((size_t)(b * SS + i)) * DD + h * HD;
        ci[lane]      = (acc0 + vs0) / denom;
        ci[lane + 32] = (acc1 + vs1) / denom;
    }
}

// ---------------------------------------------------------------------------
// Host-side orchestration
// ---------------------------------------------------------------------------
extern "C" void kernel_launch(void* const* d_in, const int* in_sizes, int n_in,
                              void* d_out, int out_size, void* d_ws, size_t ws_size,
                              hipStream_t stream) {
    (void)in_sizes; (void)n_in; (void)out_size; (void)ws_size;

    const float* hidden = (const float*)d_in[0];
    const float* Wq = (const float*)d_in[1];
    const float* Wk = (const float*)d_in[2];
    const float* Wv = (const float*)d_in[3];
    const float* Wo = (const float*)d_in[4];
    const float* bq = (const float*)d_in[5];
    const float* bk = (const float*)d_in[6];
    const float* bv = (const float*)d_in[7];
    const float* bo = (const float*)d_in[8];
    float* out = (float*)d_out;

    // Workspace layout (~72 MB):
    char* base = (char*)d_ws;
    const size_t MB = 1u << 20;
    unsigned short* wqb = (unsigned short*)(base + 0 * MB);   // 2 MB each, transposed
    unsigned short* wkb = (unsigned short*)(base + 2 * MB);
    unsigned short* wvb = (unsigned short*)(base + 4 * MB);
    unsigned short* wob = (unsigned short*)(base + 6 * MB);
    float* qb   = (float*)(base + 8 * MB);    // 16 MB each
    float* kb   = (float*)(base + 24 * MB);
    float* vb   = (float*)(base + 40 * MB);
    float* ctx  = (float*)(base + 56 * MB);
    float* vsum = (float*)(base + 72 * MB);   // 8 KB

    cvt_transpose_bf16<<<1024, 256, 0, stream>>>(Wq, wqb);
    cvt_transpose_bf16<<<1024, 256, 0, stream>>>(Wk, wkb);
    cvt_transpose_bf16<<<1024, 256, 0, stream>>>(Wv, wvb);
    cvt_transpose_bf16<<<1024, 256, 0, stream>>>(Wo, wob);

    // Fused QKV projection: x tile staged once, reused for 3 weights.
    dim3 ggrid(DD / 64, MM / 128);   // (16, 32)
    gemm_bf16_kernel<3><<<ggrid, 256, 0, stream>>>(
        hidden, wqb, wkb, wvb, bq, bk, bv, qb, kb, vb);

    vsum_kernel<<<BB * HH, 256, 0, stream>>>(vb, vsum);

    band_attn_kernel<<<dim3(SS / 64, BB * HH), 256, 0, stream>>>(
        qb, kb, vb, vsum, ctx);

    // Output projection
    gemm_bf16_kernel<1><<<ggrid, 256, 0, stream>>>(
        ctx, wob, nullptr, nullptr, bo, nullptr, nullptr, out, nullptr, nullptr);
}
